// MVP_CrossAttn_GridBias_64759516889852
// MI455X (gfx1250) — matrix operand order repaired and checked by reading, not compiled, not run on hardware
//
#include <hip/hip_runtime.h>
#include <hip/hip_bf16.h>
#include <math.h>

// ---------------- constants ----------------
#define BATCH 4
#define NMEAS 1024
#define STOT  1025            // N + NBS
#define DMODEL 128
#define NHEADS 4
#define DHEAD 32
#define GDIM 64
#define KGRID 4096            // G*G
#define FFDIM 512
#define HW 256
#define PPATCH 81
#define PPAD 96

typedef __bf16 v16bf __attribute__((ext_vector_type(16)));
typedef float  v8f   __attribute__((ext_vector_type(8)));

union BF16x16 { v16bf v; unsigned short u[16]; };

__device__ __forceinline__ unsigned short f2bf(float f) {
  unsigned int x = __float_as_uint(f);
  unsigned int r = x + 0x7FFFu + ((x >> 16) & 1u);
  return (unsigned short)(r >> 16);
}
__device__ __forceinline__ float geluf(float x) {
  return 0.5f * x * (1.0f + erff(x * 0.70710678118654752f));
}
__device__ __forceinline__ int iclampi(int v, int lo, int hi) {
  return v < lo ? lo : (v > hi ? hi : v);
}

__device__ __forceinline__ v8f wmma_bf16(v16bf a, v16bf b, v8f c) {
  return __builtin_amdgcn_wmma_f32_16x16x32_bf16(false, a, false, b, (short)0, c, false, false);
}

// A fragment: 16x32 bf16, row-major source with leading dim ld.
// lane: m = lane&15 ; half = lane>>4 ; element e -> k = half*16 + e
__device__ __forceinline__ v16bf load_a_frag(const unsigned short* __restrict__ A,
                                             int row0, int k0, int ld, int lane, int rowMax) {
  BF16x16 r;
  int m = lane & 15, half = lane >> 4;
  int row = row0 + m; if (row > rowMax) row = rowMax;
  const unsigned short* p = A + (size_t)row * ld + k0 + half * 16;
#pragma unroll
  for (int e = 0; e < 16; ++e) r.u[e] = p[e];
  return r.v;
}

// Packed B fragment: 16 contiguous shorts per lane (2 x b128)
__device__ __forceinline__ v16bf load_packed_frag(const unsigned short* __restrict__ p) {
  BF16x16 r;
#pragma unroll
  for (int e = 0; e < 16; ++e) r.u[e] = p[e];
  return r.v;
}

// K^T fragment for attention: B-matrix 32(dh) x 16(keys); K stored (rows=key) x DMODEL
__device__ __forceinline__ v16bf load_kT_frag(const unsigned short* __restrict__ Kb,
                                              size_t rowBase, int key0, int hOff, int lane) {
  BF16x16 r;
  int n = lane & 15, halfK = lane >> 4;
  int key = key0 + n;
  if (key < STOT) {
    const unsigned short* p = Kb + (rowBase + key) * DMODEL + hOff + halfK * 16;
#pragma unroll
    for (int e = 0; e < 16; ++e) r.u[e] = p[e];
  } else {
#pragma unroll
    for (int e = 0; e < 16; ++e) r.u[e] = 0;
  }
  return r.v;
}

// V fragment from V^T (layout [b][d][s]): contiguous 16 shorts per lane
__device__ __forceinline__ v16bf load_vT_frag(const unsigned short* __restrict__ vT,
                                              int bIdx, int key0, int col, int lane) {
  BF16x16 r;
  int n = lane & 15, halfK = lane >> 4;
  int kbase = key0 + halfK * 16;
  const unsigned short* p = vT + ((size_t)bIdx * DMODEL + col + n) * STOT + kbase;
  int rem = STOT - kbase;
  if (rem >= 16) {
#pragma unroll
    for (int e = 0; e < 16; ++e) r.u[e] = p[e];
  } else {
#pragma unroll
    for (int e = 0; e < 16; ++e) r.u[e] = (e < rem) ? p[e] : (unsigned short)0;
  }
  return r.v;
}

// P fragment from LDS (row-major 16x32 bf16)
__device__ __forceinline__ v16bf load_p_frag(const unsigned short* __restrict__ pst, int lane) {
  BF16x16 r;
  int m = lane & 15, half = lane >> 4;
  const unsigned short* p = pst + m * 32 + half * 16;
#pragma unroll
  for (int e = 0; e < 16; ++e) r.u[e] = p[e];
  return r.v;
}

// ---------------- generic bf16 GEMM, packed B, 32x64 wave tile ----------------
// C(MxN) = A(MxK)bf16 @ Bpacked + bias ; flags: 1=GELU, 2=Cf+=, 4=Cf+=scaleRow[r]*v
__global__ void __launch_bounds__(128)
gemm_bf16_kernel(const unsigned short* __restrict__ A, const unsigned short* __restrict__ Bp,
                 const float* __restrict__ bias, float* __restrict__ Cf,
                 unsigned short* __restrict__ Cb, int M, int Kin, int Nout, int flags,
                 const float* __restrict__ scaleRow, int wStride, int rowsPerBatch) {
  int lane = threadIdx.x & 31;
  int w = threadIdx.x >> 5;
  int row0 = blockIdx.x * 128 + w * 32;
  if (row0 >= M) return;                       // wave-uniform
  int col0 = blockIdx.y * 64;
  const unsigned short* Bq = Bp;
  const float* biasp = bias;
  if (wStride) {
    int bidx = row0 / rowsPerBatch;
    Bq += (size_t)bidx * wStride;
    if (biasp) biasp += (size_t)bidx * Nout;
  }
  int ntiles = Nout >> 4;
  v8f acc[2][4];
  v8f zero = {};
#pragma unroll
  for (int s = 0; s < 2; ++s)
#pragma unroll
    for (int t = 0; t < 4; ++t) acc[s][t] = zero;
  int rowMax = M - 1;
  for (int k0 = 0; k0 < Kin; k0 += 32) {
    v16bf a0 = load_a_frag(A, row0,      k0, Kin, lane, rowMax);
    v16bf a1 = load_a_frag(A, row0 + 16, k0, Kin, lane, rowMax);
    const unsigned short* bt = Bq + ((size_t)(k0 >> 5) * ntiles + (col0 >> 4)) * 512 + lane * 16;
    if (k0 + 32 < Kin)
      __builtin_prefetch(bt + (size_t)ntiles * 512, 0, 0);
#pragma unroll
    for (int t = 0; t < 4; ++t) {
      v16bf bf = load_packed_frag(bt + (size_t)t * 512);
      acc[0][t] = wmma_bf16(a0, bf, acc[0][t]);
      acc[1][t] = wmma_bf16(a1, bf, acc[1][t]);
    }
  }
  int n = lane & 15, half = lane >> 4;
#pragma unroll
  for (int s = 0; s < 2; ++s) {
#pragma unroll
    for (int t = 0; t < 4; ++t) {
#pragma unroll
      for (int j = 0; j < 8; ++j) {
        int r = row0 + s * 16 + j + 8 * half;
        if (r < M) {
          int c = col0 + t * 16 + n;
          float v = acc[s][t][j] + (biasp ? biasp[c] : 0.0f);
          if (flags & 1) v = geluf(v);
          size_t ci = (size_t)r * Nout + c;
          if (Cf) {
            if (flags & 4) Cf[ci] += scaleRow[r] * v;
            else if (flags & 2) Cf[ci] += v;
            else Cf[ci] = v;
          }
          if (Cb) Cb[ci] = f2bf(v);
        }
      }
    }
  }
}

// ---------------- flash cross-attention (bias computed analytically) ----------------
__global__ void __launch_bounds__(128)
attn_kernel(const unsigned short* __restrict__ Qb, const unsigned short* __restrict__ Kb,
            const unsigned short* __restrict__ vT, const float* __restrict__ srcXY,
            unsigned short* __restrict__ Ob) {
  __shared__ unsigned short pst[4][16 * 32];
  int lane = threadIdx.x & 31;
  int w = threadIdx.x >> 5;
  int tile = blockIdx.x * 4 + w;          // B*H*(K/16) = 4096 tiles
  int qt = tile & 255;
  int h = (tile >> 8) & 3;
  int bIdx = tile >> 10;
  size_t qRowBase = (size_t)bIdx * KGRID;
  size_t sRowBase = (size_t)bIdx * STOT;
  v16bf qf = load_a_frag(Qb, (int)(qRowBase + qt * 16), h * DHEAD, DMODEL, lane, 1 << 29);
  float rowM[8], rowS[8];
  v8f o0 = {}, o1 = {};
#pragma unroll
  for (int j = 0; j < 8; ++j) { rowM[j] = -3e38f; rowS[j] = 0.f; }
  int n = lane & 15, half = lane >> 4;
  const float scale = 0.17677669529663687f;  // 1/sqrt(32)
  for (int c = 0; c < 33; ++c) {
    int key0 = c * 32;
    v16bf kf0 = load_kT_frag(Kb, sRowBase, key0,      h * DHEAD, lane);
    v16bf kf1 = load_kT_frag(Kb, sRowBase, key0 + 16, h * DHEAD, lane);
    v8f z = {};
    v8f s0 = wmma_bf16(qf, kf0, z);
    v8f s1 = wmma_bf16(qf, kf1, z);
    int keyA = key0 + n, keyB = key0 + 16 + n;
    bool vA = keyA < STOT, vB = keyB < STOT;
    float sxA = 0.f, syA = 0.f, sxB = 0.f, syB = 0.f;
    if (vA) { sxA = srcXY[(sRowBase + keyA) * 2]; syA = srcXY[(sRowBase + keyA) * 2 + 1]; }
    if (vB) { sxB = srcXY[(sRowBase + keyB) * 2]; syB = srcXY[(sRowBase + keyB) * 2 + 1]; }
#pragma unroll
    for (int j = 0; j < 8; ++j) {
      int kq = qt * 16 + j + 8 * half;
      float gx = ((kq & 63) + 0.5f) * (1.0f / 64.0f);
      float gy = ((kq >> 6) + 0.5f) * (1.0f / 64.0f);
      float v0 = -3e38f, v1 = -3e38f;
      if (vA) { float dx = gx - sxA, dy = gy - syA; v0 = s0[j] * scale - sqrtf(dx * dx + dy * dy); }
      if (vB) { float dx = gx - sxB, dy = gy - syB; v1 = s1[j] * scale - sqrtf(dx * dx + dy * dy); }
      float cm = fmaxf(v0, v1);
#pragma unroll
      for (int off = 8; off > 0; off >>= 1) cm = fmaxf(cm, __shfl_xor(cm, off, 16));
      float nm = fmaxf(rowM[j], cm);
      float alpha = __expf(rowM[j] - nm);
      float p0 = vA ? __expf(v0 - nm) : 0.f;
      float p1 = vB ? __expf(v1 - nm) : 0.f;
      float ps = p0 + p1;
#pragma unroll
      for (int off = 8; off > 0; off >>= 1) ps += __shfl_xor(ps, off, 16);
      rowS[j] = rowS[j] * alpha + ps;
      rowM[j] = nm;
      o0[j] *= alpha; o1[j] *= alpha;
      pst[w][(j + 8 * half) * 32 + n] = f2bf(p0);
      pst[w][(j + 8 * half) * 32 + 16 + n] = f2bf(p1);
    }
    v16bf pf = load_p_frag(&pst[w][0], lane);
    v16bf vf0 = load_vT_frag(vT, bIdx, key0, h * DHEAD, lane);
    v16bf vf1 = load_vT_frag(vT, bIdx, key0, h * DHEAD + 16, lane);
    o0 = wmma_bf16(pf, vf0, o0);
    o1 = wmma_bf16(pf, vf1, o1);
  }
#pragma unroll
  for (int j = 0; j < 8; ++j) {
    float inv = 1.0f / rowS[j];
    size_t r = qRowBase + qt * 16 + j + 8 * half;
    Ob[r * DMODEL + h * DHEAD + n]      = f2bf(o0[j] * inv);
    Ob[r * DMODEL + h * DHEAD + 16 + n] = f2bf(o1[j] * inv);
  }
}

// ---------------- LayerNorm -> bf16 ----------------
__global__ void __launch_bounds__(128)
ln_bf16_kernel(const float* __restrict__ X, const float* __restrict__ g,
               const float* __restrict__ bta, unsigned short* __restrict__ Y) {
  int r = blockIdx.x;
  int t = threadIdx.x;
  float v = X[(size_t)r * DMODEL + t];
  __shared__ float sm[4];
  float s = v;
#pragma unroll
  for (int off = 16; off > 0; off >>= 1) s += __shfl_xor(s, off, 32);
  if ((t & 31) == 0) sm[t >> 5] = s;
  __syncthreads();
  float mean = (sm[0] + sm[1] + sm[2] + sm[3]) * (1.0f / 128.0f);
  __syncthreads();
  float d = v - mean;
  float s2 = d * d;
#pragma unroll
  for (int off = 16; off > 0; off >>= 1) s2 += __shfl_xor(s2, off, 32);
  if ((t & 31) == 0) sm[t >> 5] = s2;
  __syncthreads();
  float var = (sm[0] + sm[1] + sm[2] + sm[3]) * (1.0f / 128.0f);
  float y = d * rsqrtf(var + 1e-5f) * g[t] + bta[t];
  Y[(size_t)r * DMODEL + t] = f2bf(y);
}

// ---------------- token prep: meas_proj + patch extraction ----------------
__global__ void __launch_bounds__(128)
prep_meas_kernel(const float* __restrict__ meas_xy, const float* __restrict__ meas_v,
                 const float* __restrict__ W0, const float* __restrict__ b0,
                 const float* __restrict__ city, float* __restrict__ meas_tok,
                 unsigned short* __restrict__ patches) {
  int tok = blockIdx.x;            // B*N
  int b = tok >> 10;
  int t = threadIdx.x;
  float mx = meas_xy[(size_t)tok * 2], my = meas_xy[(size_t)tok * 2 + 1], mv = meas_v[tok];
  meas_tok[(size_t)tok * DMODEL + t] = mx * W0[t] + my * W0[128 + t] + mv * W0[256 + t] + b0[t];
  if (t < PPAD) {
    unsigned short val = 0;
    if (t < PPATCH) {
      int x = iclampi((int)roundf(mx * 255.f), 0, 255);
      int y = iclampi((int)roundf(my * 255.f), 0, 255);
      int dy = t / 9 - 4, dx = t % 9 - 4;
      int yy = y + dy, xx = x + dx;
      float cv = 0.f;
      if (yy >= 0 && yy < HW && xx >= 0 && xx < HW) cv = city[((size_t)b * HW + yy) * HW + xx];
      val = f2bf(cv);
    }
    patches[(size_t)tok * PPAD + t] = val;
  }
}

// ---------------- src assembly (+ bs token + srcXY) ----------------
__global__ void __launch_bounds__(128)
assemble_src_kernel(const float* __restrict__ meas_tok, const float* __restrict__ meas_xy,
                    const float* __restrict__ bs_xy, const float* __restrict__ Wbs,
                    const float* __restrict__ bbs, float* __restrict__ src,
                    float* __restrict__ srcXY) {
  int r = blockIdx.x;              // B*STOT
  int b = r / STOT, s = r % STOT;
  int t = threadIdx.x;
  float val, x, y;
  if (s < NMEAS) {
    val = meas_tok[((size_t)b * NMEAS + s) * DMODEL + t];
    x = meas_xy[((size_t)b * NMEAS + s) * 2];
    y = meas_xy[((size_t)b * NMEAS + s) * 2 + 1];
  } else {
    x = bs_xy[b * 2]; y = bs_xy[b * 2 + 1];
    val = x * Wbs[t] + y * Wbs[128 + t] + bbs[t];
  }
  src[(size_t)r * DMODEL + t] = val;
  if (t == 0) { srcXY[r * 2] = x; srcXY[r * 2 + 1] = y; }
}

// ---------------- density ----------------
__global__ void __launch_bounds__(128)
density_kernel(const float* __restrict__ meas_xy, float* __restrict__ density) {
  int cell = blockIdx.x;           // B*K
  int b = cell >> 12, k = cell & 4095;
  int t = threadIdx.x;
  float gx = ((k & 63) + 0.5f) * (1.0f / 64.0f);
  float gy = ((k >> 6) + 0.5f) * (1.0f / 64.0f);
  float s = 0.f;
  const float c = 1.0f / (2.0f * 0.08f * 0.08f);
  for (int i = t; i < NMEAS; i += 128) {
    float dx = gx - meas_xy[((size_t)b * NMEAS + i) * 2];
    float dy = gy - meas_xy[((size_t)b * NMEAS + i) * 2 + 1];
    s += __expf(-(dx * dx + dy * dy) * c);
  }
  __shared__ float sm[4];
#pragma unroll
  for (int off = 16; off > 0; off >>= 1) s += __shfl_xor(s, off, 32);
  if ((t & 31) == 0) sm[t >> 5] = s;
  __syncthreads();
  if (t == 0) density[cell] = (sm[0] + sm[1] + sm[2] + sm[3]) * (1.0f / 1024.0f);
}

// ---------------- deterministic rasterize (per-cell scan) ----------------
__global__ void raster_kernel(const float* __restrict__ meas_xy, const float* __restrict__ meas_v,
                              float* __restrict__ cnt, float* __restrict__ summ) {
  int idx = blockIdx.x * blockDim.x + threadIdx.x;
  if (idx >= BATCH * KGRID) return;
  int b = idx >> 12, cell = idx & 4095;
  float c = 0.f, s = 0.f;
  for (int i = 0; i < NMEAS; ++i) {
    float x = meas_xy[((size_t)b * NMEAS + i) * 2];
    float y = meas_xy[((size_t)b * NMEAS + i) * 2 + 1];
    int xg = iclampi((int)roundf(x * 63.f), 0, 63);
    int yg = iclampi((int)roundf(y * 63.f), 0, 63);
    if (yg * 64 + xg == cell) { c += 1.f; s += meas_v[b * NMEAS + i]; }
  }
  cnt[idx] = c; summ[idx] = s;
}

__global__ void __launch_bounds__(256)
cntmax_kernel(const float* __restrict__ cnt, float* __restrict__ cmax) {
  int b = blockIdx.x, t = threadIdx.x;
  float m = 0.f;
  for (int i = t; i < KGRID; i += 256) m = fmaxf(m, cnt[b * KGRID + i]);
  __shared__ float sm[8];
#pragma unroll
  for (int off = 16; off > 0; off >>= 1) m = fmaxf(m, __shfl_xor(m, off, 32));
  if ((t & 31) == 0) sm[t >> 5] = m;
  __syncthreads();
  if (t == 0) {
    m = sm[0];
    for (int i = 1; i < 8; ++i) m = fmaxf(m, sm[i]);
    cmax[b] = fmaxf(m, 1.0f);
  }
}

// ---------------- grid inputs (5 channels) ----------------
__global__ void ginputs_kernel(const float* __restrict__ city, const float* __restrict__ cnt,
                               const float* __restrict__ summ, const float* __restrict__ cmax,
                               const float* __restrict__ bs_xy, float* __restrict__ gin) {
  int idx = blockIdx.x * blockDim.x + threadIdx.x;
  if (idx >= BATCH * KGRID) return;
  int b = idx >> 12, k = idx & 4095;
  int gi = k >> 6, gj = k & 63;
  float cityv = city[((size_t)b * HW + gi * 4) * HW + gj * 4];
  float c = cnt[idx];
  float meanv = summ[idx] / fmaxf(c, 1.0f);
  float cn = c / cmax[b];
  float bx = bs_xy[b * 2], by = bs_xy[b * 2 + 1];
  int xb = iclampi((int)roundf(bx * 63.f), 0, 63);
  int yb = iclampi((int)roundf(by * 63.f), 0, 63);
  float bsm = (k == yb * 64 + xb) ? 1.0f : 0.0f;
  float gx = (gj + 0.5f) * (1.0f / 64.0f), gy = (gi + 0.5f) * (1.0f / 64.0f);
  float dist = sqrtf((gx - bx) * (gx - bx) + (gy - by) * (gy - by));
  const float c0 = 0.5f / 64.0f, c1v = 63.5f / 64.0f;
  float dmax = 0.f;
  {
    float xs[2] = {c0, c1v}, ys[2] = {c0, c1v};
#pragma unroll
    for (int a = 0; a < 2; ++a)
#pragma unroll
      for (int d2 = 0; d2 < 2; ++d2) {
        float dx = xs[a] - bx, dy = ys[d2] - by;
        dmax = fmaxf(dmax, sqrtf(dx * dx + dy * dy));
      }
    dmax = fmaxf(dmax, 1e-6f);
  }
  size_t base = (size_t)b * 5 * KGRID;
  gin[base + 0 * KGRID + k] = cityv;
  gin[base + 1 * KGRID + k] = cn;
  gin[base + 2 * KGRID + k] = meanv;
  gin[base + 3 * KGRID + k] = bsm;
  gin[base + 4 * KGRID + k] = dist / dmax;
}

// ---------------- convs ----------------
__global__ void conv1_kernel(const float* __restrict__ gin, const float* __restrict__ w,
                             const float* __restrict__ bias, float* __restrict__ out) {
  int idx = blockIdx.x * blockDim.x + threadIdx.x;   // B*32*4096
  if (idx >= BATCH * 32 * KGRID) return;
  int k = idx & 4095, co = (idx >> 12) & 31, b = idx >> 17;
  int yi = k >> 6, xi = k & 63;
  float acc = bias[co];
  for (int ci = 0; ci < 5; ++ci)
#pragma unroll
    for (int ky = 0; ky < 3; ++ky)
#pragma unroll
      for (int kx = 0; kx < 3; ++kx) {
        int yy = yi + ky - 1, xx = xi + kx - 1;
        if (yy >= 0 && yy < 64 && xx >= 0 && xx < 64)
          acc += gin[((size_t)b * 5 + ci) * KGRID + yy * 64 + xx] * w[((co * 5 + ci) * 3 + ky) * 3 + kx];
      }
  out[idx] = geluf(acc);
}

__global__ void conv2_kernel(const float* __restrict__ c1o, const float* __restrict__ w,
                             const float* __restrict__ bias, float* __restrict__ out) {
  int idx = blockIdx.x * blockDim.x + threadIdx.x;   // B*128*4096
  if (idx >= BATCH * 128 * KGRID) return;
  int k = idx & 4095, d = (idx >> 12) & 127, b = idx >> 19;
  int yi = k >> 6, xi = k & 63;
  float acc = bias[d];
  for (int ci = 0; ci < 32; ++ci)
#pragma unroll
    for (int ky = 0; ky < 3; ++ky)
#pragma unroll
      for (int kx = 0; kx < 3; ++kx) {
        int yy = yi + ky - 1, xx = xi + kx - 1;
        if (yy >= 0 && yy < 64 && xx >= 0 && xx < 64)
          acc += c1o[((size_t)b * 32 + ci) * KGRID + yy * 64 + xx] * w[((d * 32 + ci) * 3 + ky) * 3 + kx];
      }
  out[idx] = acc;
}

// ---------------- task vec / gathers / grid init ----------------
__global__ void __launch_bounds__(128)
taskvec_kernel(const int* __restrict__ task_id, const float* __restrict__ temb,
               const float* __restrict__ Wt, const float* __restrict__ bt,
               const float* __restrict__ tbAll, float* __restrict__ tvec,
               float* __restrict__ taskbg) {
  int b = blockIdx.x, t = threadIdx.x;
  int tid = task_id[b];
  float s = bt[t];
  for (int e = 0; e < 128; ++e) s += temb[tid * 128 + e] * Wt[e * 128 + t];
  tvec[b * 128 + t] = s;
  taskbg[b * 128 + t] = tbAll[tid * 128 + t];
}

__global__ void __launch_bounds__(128)
initgrid_kernel(const float* __restrict__ gpos, const float* __restrict__ tvec,
                const float* __restrict__ c2o, float* __restrict__ grid) {
  int r = blockIdx.x;              // B*K
  int b = r >> 12, k = r & 4095;
  int t = threadIdx.x;
  grid[(size_t)r * DMODEL + t] =
      gpos[(size_t)k * DMODEL + t] + tvec[b * 128 + t] + c2o[((size_t)b * 128 + t) * KGRID + k];
}

// ---------------- pack / convert ----------------
// Pack f32 KxN row-major weight into WMMA-B fragment order:
// dst[((kt*ntiles + nt)*512) + lane*16 + e] = W[kt*32 + (lane>>4)*16 + e][nt*16 + (lane&15)]
__global__ void packB_kernel(const float* __restrict__ src, unsigned short* __restrict__ dst,
                             int Kin, int Nout, int srcRows) {
  int idx = blockIdx.x * blockDim.x + threadIdx.x;
  if (idx >= Kin * Nout) return;
  int e = idx & 15;
  int lane = (idx >> 4) & 31;
  int tile = idx >> 9;
  int ntiles = Nout >> 4;
  int kt = tile / ntiles, nt = tile - kt * ntiles;
  int n = lane & 15, halfK = lane >> 4;
  int k = kt * 32 + halfK * 16 + e;
  int col = nt * 16 + n;
  float v = (k < srcRows) ? src[(size_t)k * Nout + col] : 0.f;
  dst[idx] = f2bf(v);
}

// Gather task_w[task_id[b]] and pack per batch (Kin=Nout=128, ntiles=8)
__global__ void packTaskW_kernel(const int* __restrict__ task_id, const float* __restrict__ tw,
                                 unsigned short* __restrict__ dst) {
  int idx = blockIdx.x * blockDim.x + threadIdx.x;   // B*16384
  if (idx >= BATCH * 16384) return;
  int b = idx >> 14, r = idx & 16383;
  int e = r & 15;
  int lane = (r >> 4) & 31;
  int tile = r >> 9;
  int kt = tile >> 3, nt = tile & 7;
  int n = lane & 15, halfK = lane >> 4;
  int k = kt * 32 + halfK * 16 + e;
  int col = nt * 16 + n;
  dst[idx] = f2bf(tw[(size_t)task_id[b] * 16384 + k * 128 + col]);
}

// V (B*STOT x D, bf16) -> V^T ([b][d][s], bf16)
__global__ void transposeV_kernel(const unsigned short* __restrict__ v,
                                  unsigned short* __restrict__ vT) {
  int idx = blockIdx.x * blockDim.x + threadIdx.x;   // B*D*STOT (output index)
  if (idx >= BATCH * DMODEL * STOT) return;
  int s = idx % STOT;
  int d = (idx / STOT) & 127;
  int b = idx / (STOT * DMODEL);
  vT[idx] = v[((size_t)b * STOT + s) * DMODEL + d];
}

__global__ void f2bf_kernel(const float* __restrict__ src, unsigned short* __restrict__ dst, int n) {
  int i = blockIdx.x * blockDim.x + threadIdx.x;
  if (i < n) dst[i] = f2bf(src[i]);
}

// ---------------- head ----------------
__global__ void __launch_bounds__(128)
head_kernel(const float* __restrict__ tf, const float* __restrict__ hw,
            const float* __restrict__ hb, float* __restrict__ out) {
  int r = blockIdx.x, t = threadIdx.x;
  float s = tf[(size_t)r * DMODEL + t] * hw[t];
  __shared__ float sm[4];
#pragma unroll
  for (int off = 16; off > 0; off >>= 1) s += __shfl_xor(s, off, 32);
  if ((t & 31) == 0) sm[t >> 5] = s;
  __syncthreads();
  if (t == 0) out[r] = sm[0] + sm[1] + sm[2] + sm[3] + hb[0];
}

// ---------------- host ----------------
static inline void launch_gemm(hipStream_t s, const unsigned short* A, const unsigned short* Bp,
                               const float* bias, float* Cf, unsigned short* Cb,
                               int M, int Kin, int Nout, int flags, const float* scaleRow,
                               int wStride, int rowsPerBatch) {
  dim3 g((M + 127) / 128, Nout / 64);
  gemm_bf16_kernel<<<g, 128, 0, s>>>(A, Bp, bias, Cf, Cb, M, Kin, Nout, flags, scaleRow,
                                     wStride, rowsPerBatch);
}

extern "C" void kernel_launch(void* const* d_in, const int* in_sizes, int n_in,
                              void* d_out, int out_size, void* d_ws, size_t ws_size,
                              hipStream_t stream) {
  const float* meas_xy = (const float*)d_in[0];
  const float* meas_v  = (const float*)d_in[1];
  const float* bs_xy   = (const float*)d_in[2];
  const int*   task_id = (const int*)d_in[3];
  const float* city    = (const float*)d_in[4];
  int pi = 5;
  auto F = [&]() { return (const float*)d_in[pi++]; };
  const float* mpW = F(); const float* mpB = F();
  const float* p1W = F(); const float* p1B = F();
  const float* p2W = F(); const float* p2B = F();
  const float* bsW = F(); const float* bsB = F();
  const float* temb = F();
  const float* tpW = F(); const float* tpB = F();
  const float* gpos = F();
  const float* c1W = F(); const float* c1B = F();
  const float* c2W = F(); const float* c2B = F();
  struct BlkP {
    const float *ln1g, *ln1b, *ln2g, *ln2b, *ln3g, *ln3b;
    const float *wq, *bq, *wk, *bk, *wv, *bv, *wo, *bo, *f1w, *f1b, *f2w, *f2b;
  } L[2];
  for (int l = 0; l < 2; ++l) {
    L[l].ln1g = F(); L[l].ln1b = F(); L[l].ln2g = F(); L[l].ln2b = F();
    L[l].ln3g = F(); L[l].ln3b = F();
    L[l].wq = F(); L[l].bq = F(); L[l].wk = F(); L[l].bk = F();
    L[l].wv = F(); L[l].bv = F(); L[l].wo = F(); L[l].bo = F();
    L[l].f1w = F(); L[l].f1b = F(); L[l].f2w = F(); L[l].f2b = F();
  }
  const float* twAll = F(); const float* tbAll = F();
  const float* hW = F();   const float* hB = F();
  (void)in_sizes; (void)n_in; (void)out_size; (void)ws_size;

  char* base = (char*)d_ws;
  size_t off = 0;
  auto alloc = [&](size_t bytes) -> void* {
    void* r = base + off;
    off += (bytes + 255) & ~(size_t)255;
    return r;
  };
  unsigned short* patches = (unsigned short*)alloc((size_t)BATCH * NMEAS * PPAD * 2);
  float* meas_tok = (float*)alloc((size_t)BATCH * NMEAS * DMODEL * 4);
  unsigned short* phid = (unsigned short*)alloc((size_t)BATCH * NMEAS * DMODEL * 2);
  float* src = (float*)alloc((size_t)BATCH * STOT * DMODEL * 4);
  float* srcXY = (float*)alloc((size_t)BATCH * STOT * 2 * 4);
  unsigned short* srcln = (unsigned short*)alloc((size_t)BATCH * STOT * DMODEL * 2);
  unsigned short* kbf = (unsigned short*)alloc((size_t)BATCH * STOT * DMODEL * 2);
  unsigned short* vbf = (unsigned short*)alloc((size_t)BATCH * STOT * DMODEL * 2);
  unsigned short* vTb = (unsigned short*)alloc((size_t)BATCH * DMODEL * STOT * 2);
  float* grid = (float*)alloc((size_t)BATCH * KGRID * DMODEL * 4);
  unsigned short* xln = (unsigned short*)alloc((size_t)BATCH * KGRID * DMODEL * 2);
  unsigned short* qbf = (unsigned short*)alloc((size_t)BATCH * KGRID * DMODEL * 2);
  unsigned short* abf = (unsigned short*)alloc((size_t)BATCH * KGRID * DMODEL * 2);
  unsigned short* ffh = (unsigned short*)alloc((size_t)BATCH * KGRID * FFDIM * 2);
  float* tfb = (float*)alloc((size_t)BATCH * KGRID * DMODEL * 4);
  float* density = (float*)alloc((size_t)BATCH * KGRID * 4);
  float* cnt = (float*)alloc((size_t)BATCH * KGRID * 4);
  float* summ = (float*)alloc((size_t)BATCH * KGRID * 4);
  float* cmax = (float*)alloc(BATCH * 4);
  float* gin = (float*)alloc((size_t)BATCH * 5 * KGRID * 4);
  float* c1o = (float*)alloc((size_t)BATCH * 32 * KGRID * 4);
  float* c2o = (float*)alloc((size_t)BATCH * 128 * KGRID * 4);
  float* tvec = (float*)alloc(BATCH * 128 * 4);
  float* taskbg = (float*)alloc(BATCH * 128 * 4);
  unsigned short* twg = (unsigned short*)alloc((size_t)BATCH * 16384 * 2);
  unsigned short* p1Wb = (unsigned short*)alloc((size_t)PPAD * 128 * 2);
  unsigned short* p2Wb = (unsigned short*)alloc(16384 * 2);
  unsigned short* wqb[2], *wkb[2], *wvb[2], *wob[2], *f1b_[2], *f2b_[2];
  for (int l = 0; l < 2; ++l) {
    wqb[l] = (unsigned short*)alloc(16384 * 2);
    wkb[l] = (unsigned short*)alloc(16384 * 2);
    wvb[l] = (unsigned short*)alloc(16384 * 2);
    wob[l] = (unsigned short*)alloc(16384 * 2);
    f1b_[l] = (unsigned short*)alloc(65536 * 2);
    f2b_[l] = (unsigned short*)alloc(65536 * 2);
  }

  auto pack = [&](const float* s, unsigned short* d, int Kin, int Nout, int srcRows) {
    int n = Kin * Nout;
    packB_kernel<<<(n + 255) / 256, 256, 0, stream>>>(s, d, Kin, Nout, srcRows);
  };
  // weight packing (B-fragment order, contiguous per lane)
  pack(p1W, p1Wb, PPAD, 128, PPATCH);
  pack(p2W, p2Wb, 128, 128, 128);
  for (int l = 0; l < 2; ++l) {
    pack(L[l].wq, wqb[l], 128, 128, 128); pack(L[l].wk, wkb[l], 128, 128, 128);
    pack(L[l].wv, wvb[l], 128, 128, 128); pack(L[l].wo, wob[l], 128, 128, 128);
    pack(L[l].f1w, f1b_[l], 128, 512, 128); pack(L[l].f2w, f2b_[l], 512, 128, 512);
  }
  packTaskW_kernel<<<(BATCH * 16384 + 255) / 256, 256, 0, stream>>>(task_id, twAll, twg);

  // tokens
  prep_meas_kernel<<<BATCH * NMEAS, 128, 0, stream>>>(meas_xy, meas_v, mpW, mpB, city,
                                                      meas_tok, patches);
  launch_gemm(stream, patches, p1Wb, p1B, nullptr, phid, BATCH * NMEAS, PPAD, DMODEL, 1,
              nullptr, 0, 0);
  launch_gemm(stream, phid, p2Wb, p2B, meas_tok, nullptr, BATCH * NMEAS, DMODEL, DMODEL, 2,
              nullptr, 0, 0);
  assemble_src_kernel<<<BATCH * STOT, 128, 0, stream>>>(meas_tok, meas_xy, bs_xy, bsW, bsB,
                                                        src, srcXY);
  // density + grid bias path
  density_kernel<<<BATCH * KGRID, 128, 0, stream>>>(meas_xy, density);
  raster_kernel<<<(BATCH * KGRID + 255) / 256, 256, 0, stream>>>(meas_xy, meas_v, cnt, summ);
  cntmax_kernel<<<BATCH, 256, 0, stream>>>(cnt, cmax);
  ginputs_kernel<<<(BATCH * KGRID + 255) / 256, 256, 0, stream>>>(city, cnt, summ, cmax,
                                                                  bs_xy, gin);
  conv1_kernel<<<(BATCH * 32 * KGRID + 255) / 256, 256, 0, stream>>>(gin, c1W, c1B, c1o);
  conv2_kernel<<<(BATCH * 128 * KGRID + 255) / 256, 256, 0, stream>>>(c1o, c2W, c2B, c2o);
  taskvec_kernel<<<BATCH, 128, 0, stream>>>(task_id, temb, tpW, tpB, tbAll, tvec, taskbg);
  initgrid_kernel<<<BATCH * KGRID, 128, 0, stream>>>(gpos, tvec, c2o, grid);

  const int MQ = BATCH * KGRID;   // 16384
  const int MS = BATCH * STOT;    // 4100
  for (int l = 0; l < 2; ++l) {
    // K,V from src (once per layer)
    ln_bf16_kernel<<<MS, 128, 0, stream>>>(src, L[l].ln2g, L[l].ln2b, srcln);
    launch_gemm(stream, srcln, wkb[l], L[l].bk, nullptr, kbf, MS, DMODEL, DMODEL, 0, nullptr, 0, 0);
    launch_gemm(stream, srcln, wvb[l], L[l].bv, nullptr, vbf, MS, DMODEL, DMODEL, 0, nullptr, 0, 0);
    transposeV_kernel<<<(BATCH * DMODEL * STOT + 255) / 256, 256, 0, stream>>>(vbf, vTb);
    // Q from grid
    ln_bf16_kernel<<<MQ, 128, 0, stream>>>(grid, L[l].ln1g, L[l].ln1b, xln);
    launch_gemm(stream, xln, wqb[l], L[l].bq, nullptr, qbf, MQ, DMODEL, DMODEL, 0, nullptr, 0, 0);
    // flash cross-attention
    attn_kernel<<<BATCH * NHEADS * (KGRID / 16) / 4, 128, 0, stream>>>(qbf, kbf, vTb, srcXY, abf);
    // grid += density * (attn @ Wo + bo)
    launch_gemm(stream, abf, wob[l], L[l].bo, grid, nullptr, MQ, DMODEL, DMODEL, 4, density, 0, 0);
    // FFN
    ln_bf16_kernel<<<MQ, 128, 0, stream>>>(grid, L[l].ln3g, L[l].ln3b, xln);
    launch_gemm(stream, xln, f1b_[l], L[l].f1b, nullptr, ffh, MQ, DMODEL, FFDIM, 1, nullptr, 0, 0);
    launch_gemm(stream, ffh, f2b_[l], L[l].f2b, grid, nullptr, MQ, FFDIM, DMODEL, 2, nullptr, 0, 0);
  }
  // task head: tf = gelu(grid @ task_w[b] + task_b[b]); logits = tf @ head
  f2bf_kernel<<<(MQ * DMODEL + 255) / 256, 256, 0, stream>>>(grid, xln, MQ * DMODEL);
  launch_gemm(stream, xln, twg, taskbg, tfb, nullptr, MQ, DMODEL, DMODEL, 1, nullptr,
              16384, KGRID);
  head_kernel<<<MQ, 128, 0, stream>>>(tfb, hW, hB, (float*)d_out);
}
